// Refine_Network_81862076662313
// MI455X (gfx1250) — compile-verified, matrix operand order checked by reading, not run on hardware
//
#include <hip/hip_runtime.h>
#include <hip/hip_bf16.h>
#include <math.h>

typedef __attribute__((ext_vector_type(16))) _Float16 v16h;
typedef __attribute__((ext_vector_type(8)))  float    v8f;

#define LQ    512
#define KNB   64
#define DMSA  256
#define DPAIR 128
#define DSTATE 32
#define DX    309          // 256 + 21 + 32
#define EDGEF 32
#define NRBF  36
#define MLPIN 135
#define MLPINP 160         // padded K for WMMA
#define MLPH  64
#define MLPOUT 38
#define MSTRIDE 40         // mLDS row stride (f32)

// ---------------- shared helpers ----------------
__device__ __forceinline__ float wave_sum(float v) {
    #pragma unroll
    for (int m = 16; m > 0; m >>= 1) v += __shfl_xor(v, m, 32);
    return v;
}

// Fragment-major LDS index helpers. WMMA 16-bit operand layout (ISA 7.12.2):
// lane ln: row/col = base + (ln&15), kb = (ln>>4)*8;
// halves e<8 -> K = 32c+kb+e ; e>=8 -> K = 32c+16+kb+(e-8).
// We store operands so each lane's 16 halves are contiguous (32B, aligned).
__device__ __forceinline__ int afrag_idx(int row, int k, int nchunks) {
    int w  = row >> 4;
    int c  = k >> 5, r = k & 31;
    int r2 = r & 15;
    int ln = (row & 15) + ((r2 >= 8) ? 16 : 0);
    int e  = (r2 & 7) + ((r >= 16) ? 8 : 0);
    return ((w * nchunks + c) * 32 + ln) * 16 + e;
}
__device__ __forceinline__ int bfrag_idx(int k, int n, int ntiles) {
    int c  = k >> 5, r = k & 31;
    int r2 = r & 15;
    int ln = (n & 15) + ((r2 >= 8) ? 16 : 0);
    int e  = (r2 & 7) + ((r >= 16) ? 8 : 0);
    int t  = n >> 4;
    return ((c * ntiles + t) * 32 + ln) * 16 + e;
}

// =====================================================================
// Kernel 1: node = LN( [LN(msa) | seq1hot | LN(state)] @ W_x + b_x )
// one block (64 threads) per residue
// =====================================================================
__global__ __launch_bounds__(64) void node_kernel(
    const float* __restrict__ msa, const float* __restrict__ state,
    const float* __restrict__ seq1hot,
    const float* __restrict__ g_msa, const float* __restrict__ b_msa,
    const float* __restrict__ g_state, const float* __restrict__ b_state,
    const float* __restrict__ W_x, const float* __restrict__ b_x,
    const float* __restrict__ g_node, const float* __restrict__ b_node,
    float* __restrict__ nodeW)
{
    __shared__ float x[DX];
    __shared__ float red[64], red2[64];
    const int l = blockIdx.x, t = threadIdx.x;

    float s = 0.f, s2 = 0.f;
    for (int i = t; i < DMSA; i += 64) { float v = msa[l*DMSA + i]; s += v; s2 += v*v; }
    red[t] = s; red2[t] = s2; __syncthreads();
    #pragma unroll
    for (int st = 32; st > 0; st >>= 1) {
        if (t < st) { red[t] += red[t+st]; red2[t] += red2[t+st]; }
        __syncthreads();
    }
    float mu = red[0] * (1.f/DMSA);
    float var = red2[0] * (1.f/DMSA) - mu*mu;
    float inv = rsqrtf(var + 1e-5f);
    __syncthreads();
    for (int i = t; i < DMSA; i += 64)
        x[i] = (msa[l*DMSA + i] - mu) * inv * g_msa[i] + b_msa[i];
    if (t < 21) x[DMSA + t] = seq1hot[l*21 + t];

    float sv = (t < DSTATE) ? state[l*DSTATE + t] : 0.f;
    red[t] = sv; red2[t] = sv*sv; __syncthreads();
    #pragma unroll
    for (int st = 32; st > 0; st >>= 1) {
        if (t < st) { red[t] += red[t+st]; red2[t] += red2[t+st]; }
        __syncthreads();
    }
    mu = red[0] * (1.f/DSTATE);
    var = red2[0] * (1.f/DSTATE) - mu*mu;
    inv = rsqrtf(var + 1e-5f);
    if (t < DSTATE) x[DMSA + 21 + t] = (sv - mu) * inv * g_state[t] + b_state[t];
    __syncthreads();

    float yo = 0.f;
    if (t < 32) {
        for (int k = 0; k < DX; ++k) yo += x[k] * W_x[k*32 + t];
        yo += b_x[t];
    }
    red[t] = (t < 32) ? yo : 0.f; red2[t] = (t < 32) ? yo*yo : 0.f; __syncthreads();
    #pragma unroll
    for (int st = 32; st > 0; st >>= 1) {
        if (t < st) { red[t] += red[t+st]; red2[t] += red2[t+st]; }
        __syncthreads();
    }
    mu = red[0] * (1.f/32.f);
    var = red2[0] * (1.f/32.f) - mu*mu;
    inv = rsqrtf(var + 1e-5f);
    if (t < 32) nodeW[l*32 + t] = (yo - mu) * inv * g_node[t] + b_node[t];
}

// =====================================================================
// Kernel 2: top-64 nearest neighbors per residue + ca + v1
// =====================================================================
__global__ __launch_bounds__(256) void topk_kernel(
    const float* __restrict__ xyz,
    int* __restrict__ nbrW, float* __restrict__ caW, float* __restrict__ v1W)
{
    __shared__ float d2s[LQ];
    __shared__ float rv[256];
    __shared__ int   ri[256];
    const int l = blockIdx.x, t = threadIdx.x;

    const float cax = xyz[(l*3+1)*3+0];
    const float cay = xyz[(l*3+1)*3+1];
    const float caz = xyz[(l*3+1)*3+2];
    for (int j = t; j < LQ; j += 256) {
        float dx = xyz[(j*3+1)*3+0] - cax;
        float dy = xyz[(j*3+1)*3+1] - cay;
        float dz = xyz[(j*3+1)*3+2] - caz;
        d2s[j] = dx*dx + dy*dy + dz*dz + ((j == l) ? 1e9f : 0.f);
    }
    if (t < 3) caW[l*3 + t] = xyz[(l*3+1)*3 + t];
    if (t < 9) {
        int a = t/3, d = t%3;
        v1W[l*9 + t] = xyz[(l*3+a)*3 + d] - xyz[(l*3+1)*3 + d];
    }
    __syncthreads();

    for (int it = 0; it < KNB; ++it) {
        float v0 = d2s[t], v1 = d2s[t+256];
        float bv; int bi;
        if (v0 <= v1) { bv = v0; bi = t; } else { bv = v1; bi = t+256; }
        rv[t] = bv; ri[t] = bi; __syncthreads();
        #pragma unroll
        for (int st = 128; st > 0; st >>= 1) {
            if (t < st) {
                float ov = rv[t+st]; int oi = ri[t+st];
                if (ov < rv[t] || (ov == rv[t] && oi < ri[t])) { rv[t] = ov; ri[t] = oi; }
            }
            __syncthreads();
        }
        if (t == 0) { nbrW[l*KNB + it] = ri[0]; d2s[ri[0]] = 3e38f; }
        __syncthreads();
    }
}

// =====================================================================
// Kernel 3: edge features + MLP via WMMA, fragment-major LDS operands.
// One block (128 thr / 4 waves) per residue; wave w owns neighbor rows
// [16w, 16w+16).
// LDS regions (phase-overlaid, 63232 B):
//   invF f16 4w x 5c x 32ln x 16h  (20480) | later hF f16 4w x 2c x 512 (8192)
//   prF  f16 4w x 4c x 512        (16384) | later w1F f16 5c x 4t x 512 (20480)
//   we1F f16 4c x 2t x 512        ( 8192) | later w2F f16 2c x 3t x 512 (6144)
//   efL  f32 [64][32]             ( 8192) | later mL  f32 [64][40]     (10240)
// =====================================================================
#define OFF_INV    0
#define OFF_W1     20480
#define OFF_PR     20480
#define OFF_WE1    40960
#define OFF_W2     40960
#define OFF_EF     49152
#define OFF_M      49152
#define OFF_NBR    59392
#define OFF_R      59648
#define OFF_RELHAT 59904
#define OFF_V1J    60672
#define OFF_NODE   62976
#define OFF_CA     63104
#define SMEM_BYTES 63232

__global__ __launch_bounds__(128) void edge_mlp_kernel(
    const float* __restrict__ pair,
    const float* __restrict__ g_pair, const float* __restrict__ b_pair,
    const float* __restrict__ W_e1, const float* __restrict__ b_e1,
    const float* __restrict__ g_ne1, const float* __restrict__ b_ne1,
    const float* __restrict__ W1, const float* __restrict__ b1,
    const float* __restrict__ W2, const float* __restrict__ b2,
    const float* __restrict__ Wv, const float* __restrict__ Wself,
    const float* __restrict__ nodeW, const float* __restrict__ caW,
    const float* __restrict__ v1W, const int* __restrict__ nbrW,
    float* __restrict__ out0W, float* __restrict__ out1W)
{
    __shared__ __align__(32) char smem[SMEM_BYTES];
    _Float16* invF  = (_Float16*)(smem + OFF_INV);
    _Float16* prF   = (_Float16*)(smem + OFF_PR);
    _Float16* we1F  = (_Float16*)(smem + OFF_WE1);
    _Float16* w1F   = (_Float16*)(smem + OFF_W1);
    _Float16* w2F   = (_Float16*)(smem + OFF_W2);
    float*    efL   = (float*)   (smem + OFF_EF);
    float*    mL    = (float*)   (smem + OFF_M);
    int*      nbrL  = (int*)     (smem + OFF_NBR);
    float*    rL    = (float*)   (smem + OFF_R);
    float*    rhL   = (float*)   (smem + OFF_RELHAT);
    float*    v1jL  = (float*)   (smem + OFF_V1J);
    float*    nodeL = (float*)   (smem + OFF_NODE);
    float*    caL   = (float*)   (smem + OFF_CA);

    const int l    = blockIdx.x;
    const int tid  = threadIdx.x;
    const int lane = tid & 31;
    const int wave = tid >> 5;

    // ---------------- phase 0: stage per-residue scalars + prefetch ----------------
    if (tid < KNB) {
        int jn = nbrW[l*KNB + tid];
        nbrL[tid] = jn;
        __builtin_prefetch(pair + ((size_t)l * LQ + (size_t)jn) * DPAIR, 0, 3);
    }
    if (tid < 32)  nodeL[tid] = nodeW[l*32 + tid];
    if (tid < 3)   caL[tid] = caW[l*3 + tid];
    __syncthreads();

    for (int i = tid; i < KNB*9; i += 128) v1jL[i] = v1W[nbrL[i/9]*9 + (i%9)];
    if (tid < KNB) {
        const int jn = nbrL[tid];
        float rx = caW[jn*3+0] - caL[0];
        float ry = caW[jn*3+1] - caL[1];
        float rz = caW[jn*3+2] - caL[2];
        float r = sqrtf(rx*rx + ry*ry + rz*rz + 1e-8f);
        rL[tid] = r;
        rhL[tid*3+0] = rx/r; rhL[tid*3+1] = ry/r; rhL[tid*3+2] = rz/r;
    }
    // stage W_e1 as f16 B-fragments (2 N-tiles)
    for (int i = tid; i < DPAIR*EDGEF; i += 128) {
        int k = i >> 5, n = i & 31;
        we1F[bfrag_idx(k, n, 2)] = (_Float16)W_e1[i];
    }
    __syncthreads();

    // ---------------- phase 1: inv features (A-frags, 5 chunks) + LN'd pair rows ----
    for (int i = tid; i < KNB*32; i += 128) { int k=i>>5, n=i&31; invF[afrag_idx(k, n,      5)] = (_Float16)nodeL[n]; }
    for (int i = tid; i < KNB*32; i += 128) { int k=i>>5, n=i&31; invF[afrag_idx(k, 32+n,   5)] = (_Float16)nodeW[nbrL[k]*32 + n]; }
    for (int i = tid; i < KNB*NRBF; i += 128) {
        int k = i / NRBF, n = i % NRBF;
        float c = 20.f * (float)n / 35.f;
        float d = (rL[k] - c) * ((float)NRBF / 20.f);
        invF[afrag_idx(k, 96+n, 5)] = (_Float16)__expf(-d*d);
    }
    for (int i = tid; i < KNB*3; i += 128) {
        int k = i/3, a = i%3;
        float s = 0.f;
        #pragma unroll
        for (int d = 0; d < 3; ++d) { float v = v1jL[k*9 + a*3 + d]; s += v*v; }
        invF[afrag_idx(k, 132+a, 5)] = (_Float16)sqrtf(s + 1e-8f);
    }
    for (int i = tid; i < KNB*(MLPINP-MLPIN); i += 128) {
        int k = i/25, p = i%25;
        invF[afrag_idx(k, MLPIN+p, 5)] = (_Float16)0.f;
    }
    // gathered pair rows: LN over 128 with g_pair/b_pair -> f16 A-fragments (4 chunks)
    for (int j = wave; j < KNB; j += 4) {
        const float* row = pair + ((size_t)l * LQ + (size_t)nbrL[j]) * DPAIR;
        float v[4];
        #pragma unroll
        for (int e = 0; e < 4; ++e) v[e] = row[lane*4 + e];
        float s  = v[0]+v[1]+v[2]+v[3];
        float s2 = v[0]*v[0]+v[1]*v[1]+v[2]*v[2]+v[3]*v[3];
        s = wave_sum(s); s2 = wave_sum(s2);
        float mu  = s * (1.f/DPAIR);
        float var = s2 * (1.f/DPAIR) - mu*mu;
        float inv = rsqrtf(var + 1e-5f);
        #pragma unroll
        for (int e = 0; e < 4; ++e) {
            int c = lane*4 + e;
            prF[afrag_idx(j, c, 4)] = (_Float16)((v[e] - mu) * inv * g_pair[c] + b_pair[c]);
        }
    }
    __syncthreads();

    // ---------------- phase 2: edge projection WMMA (64x128 @ 128x32) ----------------
    {
        const v16h* aV = (const v16h*)prF;
        const v16h* bV = (const v16h*)we1F;
        v8f acc[2] = {};
        for (int c = 0; c < 4; ++c) {
            v16h a = aV[(wave*4 + c)*32 + lane];
            #pragma unroll
            for (int t = 0; t < 2; ++t) {
                v16h b = bV[(c*2 + t)*32 + lane];
                acc[t] = __builtin_amdgcn_wmma_f32_16x16x32_f16(
                    false, a, false, b, (short)0, acc[t], false, false);
            }
        }
        #pragma unroll
        for (int t = 0; t < 2; ++t) {
            int col = 16*t + (lane & 15);
            #pragma unroll
            for (int j = 0; j < 8; ++j) {
                int r2 = wave*16 + j + 8*(lane >> 4);
                efL[r2*EDGEF + col] = acc[t][j] + b_e1[col];
            }
        }
    }
    __syncthreads();

    // ---------------- phase 3: LN(efeat) -> inv features [64..96) ; stage W1 --------
    if (tid < KNB) {
        int k = tid; float s = 0.f, s2 = 0.f;
        for (int n = 0; n < 32; ++n) { float v = efL[k*EDGEF + n]; s += v; s2 += v*v; }
        float mu  = s * (1.f/32.f);
        float var = s2 * (1.f/32.f) - mu*mu;
        float inv = rsqrtf(var + 1e-5f);
        for (int n = 0; n < 32; ++n)
            invF[afrag_idx(k, 64+n, 5)] =
                (_Float16)((efL[k*EDGEF + n] - mu) * inv * g_ne1[n] + b_ne1[n]);
    }
    for (int i = tid; i < MLPINP*MLPH; i += 128) {
        int k = i >> 6, n = i & 63;
        w1F[bfrag_idx(k, n, 4)] = (_Float16)((k < MLPIN) ? W1[k*MLPH + n] : 0.f);
    }
    __syncthreads();

    // ---------------- phase 4: MLP1 WMMA (64x160 @ 160x64), relu --------------------
    v8f acc1[4] = {};
    {
        const v16h* aV = (const v16h*)invF;
        const v16h* bV = (const v16h*)w1F;
        for (int c = 0; c < 5; ++c) {
            v16h a = aV[(wave*5 + c)*32 + lane];
            #pragma unroll
            for (int t = 0; t < 4; ++t) {
                v16h b = bV[(c*4 + t)*32 + lane];
                acc1[t] = __builtin_amdgcn_wmma_f32_16x16x32_f16(
                    false, a, false, b, (short)0, acc1[t], false, false);
            }
        }
    }
    __syncthreads();   // all invF reads done -> safe to overlay with hF

    _Float16* hF = (_Float16*)(smem + OFF_INV);
    #pragma unroll
    for (int t = 0; t < 4; ++t) {
        int col = 16*t + (lane & 15);
        #pragma unroll
        for (int j = 0; j < 8; ++j) {
            int r2 = wave*16 + j + 8*(lane >> 4);
            float v = acc1[t][j] + b1[col];
            hF[afrag_idx(r2, col, 2)] = (_Float16)(v > 0.f ? v : 0.f);
        }
    }
    for (int i = tid; i < MLPH*48; i += 128) {
        int k = i/48, n = i%48;
        w2F[bfrag_idx(k, n, 3)] = (_Float16)((n < MLPOUT) ? W2[k*MLPOUT + n] : 0.f);
    }
    __syncthreads();

    // ---------------- phase 5: MLP2 WMMA (64x64 @ 64x48) ----------------------------
    v8f acc2[3] = {};
    {
        const v16h* aV = (const v16h*)hF;
        const v16h* bV = (const v16h*)w2F;
        for (int c = 0; c < 2; ++c) {
            v16h a = aV[(wave*2 + c)*32 + lane];
            #pragma unroll
            for (int t = 0; t < 3; ++t) {
                v16h b = bV[(c*3 + t)*32 + lane];
                acc2[t] = __builtin_amdgcn_wmma_f32_16x16x32_f16(
                    false, a, false, b, (short)0, acc2[t], false, false);
            }
        }
    }
    __syncthreads();   // efL reads long done -> safe to overlay with mL
    #pragma unroll
    for (int t = 0; t < 3; ++t) {
        int col = 16*t + (lane & 15);
        if (col < MSTRIDE) {
            #pragma unroll
            for (int j = 0; j < 8; ++j) {
                int r2 = wave*16 + j + 8*(lane >> 4);
                mL[r2*MSTRIDE + col] = acc2[t][j] + ((col < MLPOUT) ? b2[col] : 0.f);
            }
        }
    }
    __syncthreads();

    // ---------------- epilogue ----------------
    if (tid < 32) {
        const int o = tid;
        float s = 0.f;
        for (int k = 0; k < KNB; ++k) s += mL[k*MSTRIDE + o];
        float dot = 0.f;
        for (int i = 0; i < 32; ++i) dot += nodeL[i] * Wself[i*32 + o];
        out0W[l*32 + o] = dot + s * (1.f/KNB);
    }
    if (tid < 9) {
        const int o = tid/3, d = tid%3;
        const float wv0 = Wv[o*3+0], wv1 = Wv[o*3+1], wv2 = Wv[o*3+2];
        float s = 0.f;
        for (int k = 0; k < KNB; ++k) {
            float vm = v1jL[k*9 + 0*3 + d]*wv0 + v1jL[k*9 + 1*3 + d]*wv1 + v1jL[k*9 + 2*3 + d]*wv2;
            s += mL[k*MSTRIDE + 32 + o]*vm + mL[k*MSTRIDE + 35 + o]*rhL[k*3 + d];
        }
        out1W[l*9 + o*3 + d] = s * (1.f/KNB);
    }
}

// =====================================================================
// Kernel 4: finalize. One wave per residue: lddt head + xyz assembly.
// =====================================================================
__global__ __launch_bounds__(32) void finalize_kernel(
    const float* __restrict__ out0W, const float* __restrict__ out1W,
    const float* __restrict__ caW,
    const float* __restrict__ g_state, const float* __restrict__ b_state,
    const float* __restrict__ W_lddt, const float* __restrict__ b_lddt,
    float* __restrict__ d_out)
{
    const int l = blockIdx.x, lane = threadIdx.x;
    float v = out0W[l*32 + lane];
    float mu = wave_sum(v) * (1.f/32.f);
    float d = v - mu;
    float var = wave_sum(d*d) * (1.f/32.f);
    float y = d * rsqrtf(var + 1e-5f) * g_state[lane] + b_state[lane];
    float dot = wave_sum(y * W_lddt[lane]);
    if (lane == 0)
        d_out[LQ*3*3 + l] = 1.f / (1.f + __expf(-(dot + b_lddt[0])));
    if (lane < 9) {
        const int a = lane/3, dd = lane%3;
        float ca_new = caW[l*3 + dd] + out1W[l*9 + 3 + dd];  // out1[:, :, 1]
        float base = (l == 0) ? 0.f : ca_new;
        d_out[(l*3 + a)*3 + dd] = out1W[l*9 + lane] + base;
    }
}

// =====================================================================
// Host launch
// =====================================================================
extern "C" void kernel_launch(void* const* d_in, const int* in_sizes, int n_in,
                              void* d_out, int out_size, void* d_ws, size_t ws_size,
                              hipStream_t stream) {
    const float* xyz     = (const float*)d_in[0];
    const float* state   = (const float*)d_in[1];
    const float* msa     = (const float*)d_in[2];
    const float* pair    = (const float*)d_in[3];
    const float* seq1hot = (const float*)d_in[4];
    // d_in[5]=idx, d_in[6]=CA_atom_index, d_in[7]=top_k : unused (top_k fixed at 64)
    const float* g_msa   = (const float*)d_in[8];
    const float* b_msa   = (const float*)d_in[9];
    const float* g_pair  = (const float*)d_in[10];
    const float* b_pair  = (const float*)d_in[11];
    const float* g_state = (const float*)d_in[12];
    const float* b_state = (const float*)d_in[13];
    const float* W_x     = (const float*)d_in[14];
    const float* b_x     = (const float*)d_in[15];
    const float* g_node  = (const float*)d_in[16];
    const float* b_node  = (const float*)d_in[17];
    const float* W_e1    = (const float*)d_in[18];
    const float* b_e1    = (const float*)d_in[19];
    const float* g_ne1   = (const float*)d_in[20];
    const float* b_ne1   = (const float*)d_in[21];
    const float* W1      = (const float*)d_in[22];
    const float* b1      = (const float*)d_in[23];
    const float* W2      = (const float*)d_in[24];
    const float* b2      = (const float*)d_in[25];
    const float* Wv      = (const float*)d_in[26];
    const float* Wself   = (const float*)d_in[27];
    const float* W_lddt  = (const float*)d_in[28];
    const float* b_lddt  = (const float*)d_in[29];

    char* ws = (char*)d_ws;
    float* nodeW = (float*)(ws + 0);        // 512*32  f32
    float* caW   = (float*)(ws + 65536);    // 512*3   f32
    float* v1W   = (float*)(ws + 71680);    // 512*9   f32
    int*   nbrW  = (int*)  (ws + 90112);    // 512*64  i32
    float* out0W = (float*)(ws + 221184);   // 512*32  f32
    float* out1W = (float*)(ws + 286720);   // 512*9   f32  (end 305152 B)

    node_kernel<<<LQ, 64, 0, stream>>>(msa, state, seq1hot, g_msa, b_msa,
                                       g_state, b_state, W_x, b_x, g_node, b_node, nodeW);
    topk_kernel<<<LQ, 256, 0, stream>>>(xyz, nbrW, caW, v1W);
    edge_mlp_kernel<<<LQ, 128, 0, stream>>>(pair, g_pair, b_pair, W_e1, b_e1,
                                            g_ne1, b_ne1, W1, b1, W2, b2, Wv, Wself,
                                            nodeW, caW, v1W, nbrW, out0W, out1W);
    finalize_kernel<<<LQ, 32, 0, stream>>>(out0W, out1W, caW, g_state, b_state,
                                           W_lddt, b_lddt, (float*)d_out);
}
